// GraphAttentionLayer_214748365433
// MI455X (gfx1250) — matrix operand order repaired and checked by reading, not compiled, not run on hardware
//
#include <hip/hip_runtime.h>
#include <cstdint>
#include <cstddef>

#define N_NODES 8192
#define F_IN    512
#define F_OUT   256
#define GAT_ALPHA 0.2f
#define GAT_NEG_INF -9000000000000000.0f

typedef __attribute__((ext_vector_type(8)))  float  v8f;
typedef __attribute__((ext_vector_type(2)))  float  v2f;
typedef __attribute__((ext_vector_type(16))) __bf16 v16bf;

static __device__ __forceinline__ unsigned short f2bf(float f) {
  unsigned int u = __float_as_uint(f);
  unsigned int r = u + 0x7FFFu + ((u >> 16) & 1u);   // round-to-nearest-even
  return (unsigned short)(r >> 16);
}

// ---------------------------------------------------------------------------
// Kernel 1: h = x @ W  (fp32, V_WMMA_F32_16X16X4_F32), one wave per 16x16 tile.
// Also writes hT in bf16 (column-major, [F_OUT][N]) for attention B-fragments.
// ---------------------------------------------------------------------------
#if __has_builtin(__builtin_amdgcn_wmma_f32_16x16x4_f32)
#define HAS_WMMA_F32 1
__global__ __launch_bounds__(128) void gemm_h_wmma(const float* __restrict__ x,
                                                   const float* __restrict__ W,
                                                   float* __restrict__ h,
                                                   unsigned short* __restrict__ hTb) {
  int lane = threadIdx.x & 31;
  int wid  = blockIdx.x * (blockDim.x >> 5) + (threadIdx.x >> 5); // 0..8191
  int tileM = wid >> 4;        // 512 row tiles
  int tileN = wid & 15;        // 16 col tiles
  int row0 = tileM * 16, col0 = tileN * 16;
  int m  = lane & 15;          // A: M index / B: N index
  int hi = lane >> 4;          // lane half selects K pair

  v8f c = {};
  const float* xp = x + (size_t)(row0 + m) * F_IN + hi * 2;        // A: lane holds K=hi*2, hi*2+1
  const float* wp = W + (size_t)(hi * 2) * F_OUT + col0 + m;       // B: lane holds K rows hi*2, hi*2+1

  for (int k = 0; k < F_IN; k += 4) {
    v2f A; A[0] = xp[k];                 A[1] = xp[k + 1];
    v2f B; B[0] = wp[(size_t)k * F_OUT]; B[1] = wp[(size_t)(k + 1) * F_OUT];
    c = __builtin_amdgcn_wmma_f32_16x16x4_f32(false, A, false, B, (short)0, c, false, false);
  }
#pragma unroll
  for (int r = 0; r < 8; ++r) {          // C: vgpr r -> M = r + 8*hi, N = lane&15
    int rr = row0 + r + hi * 8;
    int cc = col0 + m;
    float v = c[r];
    h[(size_t)rr * F_OUT + cc] = v;
    hTb[(size_t)cc * N_NODES + rr] = f2bf(v);
  }
}
#else
#define HAS_WMMA_F32 0
__global__ __launch_bounds__(256) void gemm_h_naive(const float* __restrict__ x,
                                                    const float* __restrict__ W,
                                                    float* __restrict__ h,
                                                    unsigned short* __restrict__ hTb) {
  int idx = blockIdx.x * blockDim.x + threadIdx.x;  // N*F_OUT threads
  int row = idx >> 8, col = idx & 255;
  float s = 0.0f;
  for (int k = 0; k < F_IN; ++k) s += x[(size_t)row * F_IN + k] * W[(size_t)k * F_OUT + col];
  h[(size_t)row * F_OUT + col] = s;
  hTb[(size_t)col * N_NODES + row] = f2bf(s);
}
#endif

// ---------------------------------------------------------------------------
// Kernel 2: f1[i] = h[i] . a[0:256],  f2[i] = h[i] . a[256:512]  (wave per row)
// ---------------------------------------------------------------------------
__global__ __launch_bounds__(256) void proj_f(const float* __restrict__ h,
                                              const float* __restrict__ a,
                                              float* __restrict__ f1,
                                              float* __restrict__ f2) {
  int lane = threadIdx.x & 31;
  int wave = threadIdx.x >> 5;
  int row  = blockIdx.x * 8 + wave;
  const float* hp = h + (size_t)row * F_OUT;
  float s1 = 0.0f, s2 = 0.0f;
  for (int k = lane; k < F_OUT; k += 32) {
    float hv = hp[k];
    s1 += hv * a[k];
    s2 += hv * a[F_OUT + k];
  }
  for (int off = 16; off > 0; off >>= 1) {
    s1 += __shfl_xor(s1, off, 32);
    s2 += __shfl_xor(s2, off, 32);
  }
  if (lane == 0) { f1[row] = s1; f2[row] = s2; }
}

// ---------------------------------------------------------------------------
// Kernel 3: per-row softmax stats (max, sum of exp) over masked scores.
// One block per row, online per-thread softmax, LDS tree combine.
// ---------------------------------------------------------------------------
__global__ __launch_bounds__(256) void row_stats(const int* __restrict__ adj,
                                                 const float* __restrict__ f1,
                                                 const float* __restrict__ f2,
                                                 float* __restrict__ rowmax,
                                                 float* __restrict__ rowsum) {
  __shared__ float sm[256];
  __shared__ float sl[256];
  int i = blockIdx.x;
  int tid = threadIdx.x;
  float f1v = f1[i];
  size_t base = (size_t)i * N_NODES;
  float m = -INFINITY, l = 0.0f;
  for (int j = tid; j < N_NODES; j += 256) {
    int av = adj[base + j];
    float tt = f1v + f2[j];
    float s = tt > 0.0f ? tt : GAT_ALPHA * tt;
    s = (av > 0) ? s : GAT_NEG_INF;
    if (s > m) { l = l * __expf(m - s) + 1.0f; m = s; }
    else       { l += __expf(s - m); }
  }
  sm[tid] = m; sl[tid] = l;
  __syncthreads();
  for (int off = 128; off > 0; off >>= 1) {
    if (tid < off) {
      float m1 = sm[tid], l1 = sl[tid];
      float m2 = sm[tid + off], l2 = sl[tid + off];
      float M = fmaxf(m1, m2);
      sm[tid] = M;
      sl[tid] = l1 * __expf(m1 - M) + l2 * __expf(m2 - M);
    }
    __syncthreads();
  }
  if (tid == 0) { rowmax[i] = sm[0]; rowsum[i] = sl[0]; }
}

// ---------------------------------------------------------------------------
// Kernel 4: out = softmax(scores) @ h  via v_wmma_f32_16x16x32_bf16.
// Block = 64 rows (4 row-tiles) x 256 cols, 8 waves. Per 256-j chunk each wave
// builds 4 P fragments (16x32 bf16, fp32 exp) at its private j-window, stages
// them in LDS (32 KB), then runs 64 WMMAs/chunk with each B fragment loaded
// once and reused across the 4 row-tiles (4x less L2 traffic for hT).
// ---------------------------------------------------------------------------
__global__ __launch_bounds__(256) void attn_wmma(const int* __restrict__ adj,
                                                 const unsigned short* __restrict__ hTb,
                                                 const float* __restrict__ f1,
                                                 const float* __restrict__ f2,
                                                 const float* __restrict__ rowmax,
                                                 const float* __restrict__ rowsum,
                                                 float* __restrict__ out) {
  __shared__ __align__(16) unsigned int pfr[4][8][32][8];  // [rowtile][subchunk][lane][dword]

  int lane = threadIdx.x & 31;
  int wave = threadIdx.x >> 5;
  int hi   = lane >> 4;
  int m    = lane & 15;
  int r0   = blockIdx.x * 64;

  float f1v[4], mv[4], invL[4];
  size_t adjrow[4];
#pragma unroll
  for (int t = 0; t < 4; ++t) {
    int row = r0 + t * 16 + m;
    f1v[t]  = f1[row];
    mv[t]   = rowmax[row];
    invL[t] = 1.0f / rowsum[row];
    adjrow[t] = (size_t)row * N_NODES;
  }

  v8f acc[4][2];
#pragma unroll
  for (int t = 0; t < 4; ++t) { acc[t][0] = (v8f)0; acc[t][1] = (v8f)0; }

  int n0 = wave * 32;
  const unsigned short* hpA = hTb + (size_t)(n0 + m) * N_NODES;        // tile0 column
  const unsigned short* hpB = hTb + (size_t)(n0 + 16 + m) * N_NODES;   // tile1 column

  for (int chunk = 0; chunk < N_NODES / 256; ++chunk) {
    int jbase = chunk * 256;
    int j0 = jbase + wave * 32;        // this wave's P j-window

    // ---- P phase: 4 fragments (one per row-tile) ----
#pragma unroll
    for (int t = 0; t < 4; ++t) {
      // keep the adj HBM stream ahead: prefetch next chunk's slice
      __builtin_prefetch(adj + adjrow[t] + j0 + 256, 0, 1);
      unsigned int packed[8];
#pragma unroll
      for (int g = 0; g < 2; ++g) {
        // A-frag K layout: lanes 0-15 -> K=0..15 ; lanes 16-31 -> K=16..31
        int kb = g * 8 + hi * 16;
        const int4* ap = (const int4*)(adj + adjrow[t] + j0 + kb);
        int4 a0 = ap[0], a1 = ap[1];
        const float4* fp = (const float4*)(f2 + j0 + kb);
        float4 q0 = fp[0], q1 = fp[1];
        float ff[8] = {q0.x, q0.y, q0.z, q0.w, q1.x, q1.y, q1.z, q1.w};
        int   av[8] = {a0.x, a0.y, a0.z, a0.w, a1.x, a1.y, a1.z, a1.w};
        float pv[8];
#pragma unroll
        for (int u = 0; u < 8; ++u) {
          float tt = f1v[t] + ff[u];
          float s  = tt > 0.0f ? tt : GAT_ALPHA * tt;
          s = (av[u] > 0) ? s : GAT_NEG_INF;
          pv[u] = __expf(s - mv[t]) * invL[t];
        }
#pragma unroll
        for (int d = 0; d < 4; ++d)
          packed[g * 4 + d] = (unsigned int)f2bf(pv[2 * d]) |
                              ((unsigned int)f2bf(pv[2 * d + 1]) << 16);
      }
      uint4 lo  = {packed[0], packed[1], packed[2], packed[3]};
      uint4 hiq = {packed[4], packed[5], packed[6], packed[7]};
      *(uint4*)&pfr[t][wave][lane][0] = lo;
      *(uint4*)&pfr[t][wave][lane][4] = hiq;
    }
    __syncthreads();

    // ---- MMA phase: B loaded once per subchunk, reused over 4 row-tiles ----
#pragma unroll
    for (int sub = 0; sub < 8; ++sub) {
      union { v16bf v; uint4 q[2]; } B0, B1;
      // B-frag: lane = N col, lanes 0-15 cover K=0..15, lanes 16-31 cover K=16..31
      int js = jbase + sub * 32 + hi * 16;
      const uint4* bpA = (const uint4*)(hpA + js);
      B0.q[0] = bpA[0]; B0.q[1] = bpA[1];
      const uint4* bpB = (const uint4*)(hpB + js);
      B1.q[0] = bpB[0]; B1.q[1] = bpB[1];
#pragma unroll
      for (int t = 0; t < 4; ++t) {
        union { v16bf v; uint4 q[2]; } A;
        A.q[0] = *(const uint4*)&pfr[t][sub][lane][0];
        A.q[1] = *(const uint4*)&pfr[t][sub][lane][4];
        acc[t][0] = __builtin_amdgcn_wmma_f32_16x16x32_bf16(false, A.v, false, B0.v, (short)0, acc[t][0], false, false);
        acc[t][1] = __builtin_amdgcn_wmma_f32_16x16x32_bf16(false, A.v, false, B1.v, (short)0, acc[t][1], false, false);
      }
    }
    __syncthreads();
  }

#pragma unroll
  for (int t = 0; t < 4; ++t) {
#pragma unroll
    for (int r = 0; r < 8; ++r) {        // C layout: vgpr r -> M = r + 8*hi, N = lane&15
      int rr = r0 + t * 16 + r + hi * 8;
      out[(size_t)rr * F_OUT + n0 + m]      = acc[t][0][r];
      out[(size_t)rr * F_OUT + n0 + 16 + m] = acc[t][1][r];
    }
  }
}

// ---------------------------------------------------------------------------
extern "C" void kernel_launch(void* const* d_in, const int* in_sizes, int n_in,
                              void* d_out, int out_size, void* d_ws, size_t ws_size,
                              hipStream_t stream) {
  (void)in_sizes; (void)n_in; (void)out_size; (void)ws_size;
  const float* x   = (const float*)d_in[0];
  const int*   adj = (const int*)d_in[1];
  const float* W   = (const float*)d_in[2];
  const float* a   = (const float*)d_in[3];
  float* out = (float*)d_out;

  char* ws = (char*)d_ws;
  float*          h    = (float*)ws;                                // 8 MB
  unsigned short* hTb  = (unsigned short*)(ws + 8388608);           // 4 MB (bf16, transposed)
  float*          f1   = (float*)(ws + 12582912);                   // 32 KB
  float*          f2   = (float*)(ws + 12615680);                   // 32 KB
  float*          rmax = (float*)(ws + 12648448);                   // 32 KB
  float*          rsum = (float*)(ws + 12681216);                   // 32 KB

#if HAS_WMMA_F32
  gemm_h_wmma<<<2048, 128, 0, stream>>>(x, W, h, hTb);              // 8192 tile-waves
#else
  gemm_h_naive<<<(N_NODES * F_OUT) / 256, 256, 0, stream>>>(x, W, h, hTb);
#endif
  proj_f<<<N_NODES / 8, 256, 0, stream>>>(h, a, f1, f2);
  row_stats<<<N_NODES, 256, 0, stream>>>(adj, f1, f2, rmax, rsum);
  attn_wmma<<<N_NODES / 64, 256, 0, stream>>>(adj, hTb, f1, f2, rmax, rsum, out);
}
